// PokerFusedEmbedding_54881092108297
// MI455X (gfx1250) — matrix operand order repaired and checked by reading, not compiled, not run on hardware
//
#include <hip/hip_runtime.h>
#include <hip/hip_bf16.h>

// MI455X / gfx1250, wave32.
// Roofline: ~140MB traffic -> ~6us @ 23.3TB/s (HBM-bound target).
// MLPs run on v_wmma_f32_16x16x32_f16 (f32 accumulate). Two-pass scheme:
//   pass1 computes LayerNorm stats (WMMA results consumed immediately),
//   pass2 recomputes WMMA per channel-tile and fuses gather+normalize+store.
// This keeps peak live VGPRs low (no [16][8] f32 arrays) for high occupancy.

#define NBB    16
#define DMODEL 256
#define BATCH  128
#define SEQ    1024
#define PADID  76

typedef __attribute__((ext_vector_type(16))) _Float16 v16h;
typedef __attribute__((ext_vector_type(8)))  _Float16 v8h;
typedef __attribute__((ext_vector_type(8)))  float    v8f;
typedef __attribute__((ext_vector_type(4)))  float    v4f;   // clang-native for NT stores

struct Params {
  const int   *ids, *streets, *ranks, *suits, *actors;
  const float *legal_masks, *ctx_feat;
  const float *base_emb, *street_emb, *rank_emb, *suit_emb, *actor_emb, *atype_emb;
  const float *legal_W, *legal_b, *legal_g, *legal_be;
  const float *cls_W,   *cls_b,   *cls_g,   *cls_be;
  const float *ctx_W,   *ctx_b,   *ctx_g,   *ctx_be;
  float       *out;
};

__device__ __forceinline__ void ld8(const float* p, float o[8]) {
  v4f a = *(const v4f*)p;
  v4f b = *(const v4f*)(p + 4);
  o[0]=a.x; o[1]=a.y; o[2]=a.z; o[3]=a.w;
  o[4]=b.x; o[5]=b.y; o[6]=b.z; o[7]=b.w;
}

// B operand (K x N=16 tokens, f16, K=16 real + 16 zero-pad).
// Lane n<16 holds token n's K=0..7 in elems 0..7, lane n+16 holds K=8..15; elems 8..15 (K=16..31) zero.
__device__ __forceinline__ v16h make_B(const float* x16, int lane) {
  float xv[16];
  ld8(x16, xv);
  ld8(x16 + 8, xv + 8);
  const bool lo = lane < 16;
  v16h bm;
#pragma unroll
  for (int i = 0; i < 16; ++i) bm[i] = (i < 8) ? (_Float16)(lo ? xv[i] : xv[(i + 8) & 15]) : (_Float16)0.f;
  return bm;
}

// One 16ch x 16tok WMMA for channel-tile t. A from LDS (transposed f16 weights):
// lanes 0-15 supply K=0..7 chunk of channel row, lanes 16-31 supply K=8..15.
__device__ __forceinline__ v8f wmma_tile(const _Float16 (*WA)[16], int t,
                                         int row, int koff, v16h bm) {
  v8h chunk = *(const v8h*)&WA[t * 16 + row][koff];
  v16h a;
#pragma unroll
  for (int i = 0; i < 8; ++i) { a[i] = chunk[i]; a[i + 8] = (_Float16)0.f; }
  v8f c = {};
  return __builtin_amdgcn_wmma_f32_16x16x32_f16(false, a, false, bm,
                                                (short)0, c, false, false);
}

// Pass 1: LayerNorm stats over this lane's token (channels split across lane^16).
__device__ __forceinline__ void mlp_stats(const _Float16 (*WA)[16], const float* Pb,
                                          v16h bm, int lane, float& mu, float& rs) {
  const int row  = lane & 15;
  const int koff = (lane >> 4) * 8;
  const int g8   = (lane >> 4) * 8;
  float ps = 0.f, psq = 0.f;
#pragma unroll
  for (int t = 0; t < 16; ++t) {
    v8f c = wmma_tile(WA, t, row, koff, bm);
    const int cb = t * 16 + g8;
#pragma unroll
    for (int r = 0; r < 8; ++r) {
      float hv = c[r] + Pb[cb + r];
      ps += hv; psq += hv * hv;
    }
  }
  ps  += __shfl_xor(ps, 16, 32);
  psq += __shfl_xor(psq, 16, 32);
  mu = ps * (1.f / 256.f);
  const float var = psq * (1.f / 256.f) - mu * mu;
  rs = rsqrtf(var + 1e-5f);
}

// Pass 2: recompute tile t, normalize with (mu,rs), ReLU, masked-accumulate into val.
__device__ __forceinline__ void mlp_apply(const _Float16 (*WA)[16], const float* Pb,
                                          const float* Pg, const float* Pbe,
                                          v16h bm, float m, float mu, float rs,
                                          int t, int lane, float val[8]) {
  const int row  = lane & 15;
  const int koff = (lane >> 4) * 8;
  const int g8   = (lane >> 4) * 8;
  v8f c = wmma_tile(WA, t, row, koff, bm);
  const int cb = t * 16 + g8;
#pragma unroll
  for (int r = 0; r < 8; ++r) {
    float o = (c[r] + Pb[cb + r] - mu) * rs * Pg[cb + r] + Pbe[cb + r];
    o = fmaxf(o, 0.f);
    val[r] = fmaf(m, o, val[r]);
  }
}

__global__ __launch_bounds__(256, 1)
void poker_fused_embedding_kernel(Params p)
{
  // LDS: 3 transposed f16 weight sets (legal/ctx/cls) + 3 (b,g,be) param sets = 33KB.
  __shared__ _Float16 sWA[3][DMODEL][16];
  __shared__ float    sP[3][3][DMODEL];

  const int tid = threadIdx.x;  // 0..255 == channel index for the fill
#pragma unroll
  for (int k = 0; k < 16; ++k) {
    sWA[0][tid][k] = (_Float16)p.legal_W[k * DMODEL + tid];
    sWA[1][tid][k] = (_Float16)p.ctx_W[k * DMODEL + tid];
    sWA[2][tid][k] = (k < 3) ? (_Float16)p.cls_W[k * DMODEL + tid] : (_Float16)0.f;
  }
  sP[0][0][tid] = p.legal_b[tid]; sP[0][1][tid] = p.legal_g[tid]; sP[0][2][tid] = p.legal_be[tid];
  sP[1][0][tid] = p.ctx_b[tid];   sP[1][1][tid] = p.ctx_g[tid];   sP[1][2][tid] = p.ctx_be[tid];
  sP[2][0][tid] = p.cls_b[tid];   sP[2][1][tid] = p.cls_g[tid];   sP[2][2][tid] = p.cls_be[tid];
  __syncthreads();

  const int wave = tid >> 5;
  const int lane = tid & 31;
  const int tile = blockIdx.x * 8 + wave;            // 8192 tiles total
  if (tile >= (BATCH * SEQ) / 16) return;
  const int b    = tile >> 6;                        // 64 tiles per sequence
  const int srow = tile & 63;
  const int s0   = srow * 16;
  const int n    = lane & 15;                        // token within tile
  const int g8   = (lane >> 4) * 8;                  // channel sub-block
  const long tok = (long)b * SEQ + s0 + n;

  // Per-token scalars (lanes L and L+16 duplicate the same token: cached).
  int id = p.ids[tok];
  const bool pad = id < 0;
  if (pad) id = PADID;
  const int street = p.streets[tok];
  int rank  = p.ranks[tok];  rank  = rank  < 0 ? 0 : (rank  > 12 ? 12 : rank);
  int suit  = p.suits[tok];  suit  = suit  < 0 ? 0 : (suit  > 3  ? 3  : suit);
  int actor = p.actors[tok]; actor = actor < 0 ? 0 : (actor > 1  ? 1  : actor);
  const bool cardm = (id >= 8)  && (id < 60);
  const bool actm  = (id >= 60) && (id < 76);
  const bool ctxm  = (id == 1);
  int aid = id - 60; aid = aid < 0 ? 0 : (aid > 15 ? 15 : aid);
  const float fcard = cardm ? 1.f : 0.f;
  const float fact  = actm  ? 1.f : 0.f;
  const float fctx  = ctxm  ? 1.f : 0.f;
  const float fcls  = (n == 0) ? 1.f : 0.f;          // s == 0 within a cls tile
  const float fkeep = pad ? 0.f : 1.f;

  const float* baseR = p.base_emb   + id     * DMODEL;
  const float* stR   = p.street_emb + street * DMODEL;
  const float* rkR   = p.rank_emb   + rank   * DMODEL;
  const float* suR   = p.suit_emb   + suit   * DMODEL;
  const float* acR   = p.actor_emb  + actor  * DMODEL;
  const float* atR   = p.atype_emb  + aid    * DMODEL;

  // Wave-uniform activity flags.
  const bool hasAct  = (__ballot(actm) != 0ull);     // ~97% of tiles
  const bool hasCtx  = (__ballot(ctxm) != 0ull);     // ~18% of tiles
  const bool clsTile = (srow == 0);                  // 128 of 8192 tiles

  // ---- Pass 1: build B operands once per tile, compute LayerNorm stats. ----
  v16h bmL = {}, bmC = {};
  float muL = 0.f, rsL = 0.f, muC = 0.f, rsC = 0.f, muS = 0.f, rsS = 0.f;
  if (hasAct) {
    bmL = make_B(p.legal_masks + ((long)b * SEQ + s0 + n) * NBB, lane);
    mlp_stats(sWA[0], sP[0][0], bmL, lane, muL, rsL);
  }
  if (hasCtx || clsTile) {
    bmC = make_B(p.ctx_feat + ((long)b * SEQ + s0 + n) * 16, lane);
    if (hasCtx)  mlp_stats(sWA[1], sP[1][0], bmC, lane, muC, rsC);
    if (clsTile) mlp_stats(sWA[2], sP[2][0], bmC, lane, muS, rsS);
  }

  // ---- Pass 2: fused gather + WMMA-normalize + store, streamed per tile. ----
  float* outTok = p.out + tok * DMODEL;
#pragma unroll 4
  for (int t = 0; t < 16; ++t) {
    const int ch = t * 16 + g8;
    float val[8], e0[8], e1[8];

    ld8(baseR + ch, val);
    ld8(stR + ch, e0);
    ld8(rkR + ch, e1);
#pragma unroll
    for (int r = 0; r < 8; ++r) val[r] += e0[r];
    ld8(suR + ch, e0);
#pragma unroll
    for (int r = 0; r < 8; ++r) val[r] = fmaf(fcard, e1[r] + e0[r], val[r]);

    if (hasAct) {
      ld8(acR + ch, e0);
      ld8(atR + ch, e1);
#pragma unroll
      for (int r = 0; r < 8; ++r) val[r] = fmaf(fact, e0[r] + e1[r], val[r]);
      mlp_apply(sWA[0], sP[0][0], sP[0][1], sP[0][2], bmL, fact, muL, rsL, t, lane, val);
    }
    if (hasCtx)
      mlp_apply(sWA[1], sP[1][0], sP[1][1], sP[1][2], bmC, fctx, muC, rsC, t, lane, val);
    if (clsTile)
      mlp_apply(sWA[2], sP[2][0], sP[2][1], sP[2][2], bmC, fcls, muS, rsS, t, lane, val);

    v4f o0, o1;
    o0.x = val[0] * fkeep; o0.y = val[1] * fkeep;
    o0.z = val[2] * fkeep; o0.w = val[3] * fkeep;
    o1.x = val[4] * fkeep; o1.y = val[5] * fkeep;
    o1.z = val[6] * fkeep; o1.w = val[7] * fkeep;
    // Output is write-once (128MB): non-temporal keeps L2 for tables/features.
    __builtin_nontemporal_store(o0, (v4f*)(outTok + ch));
    __builtin_nontemporal_store(o1, (v4f*)(outTok + ch + 4));
  }
}

extern "C" void kernel_launch(void* const* d_in, const int* in_sizes, int n_in,
                              void* d_out, int out_size, void* d_ws, size_t ws_size,
                              hipStream_t stream) {
  Params p;
  p.ids         = (const int*)d_in[0];
  p.streets     = (const int*)d_in[1];
  p.ranks       = (const int*)d_in[2];
  p.suits       = (const int*)d_in[3];
  p.actors      = (const int*)d_in[4];
  p.legal_masks = (const float*)d_in[5];
  p.ctx_feat    = (const float*)d_in[6];
  p.base_emb    = (const float*)d_in[7];
  p.street_emb  = (const float*)d_in[8];
  p.rank_emb    = (const float*)d_in[9];
  p.suit_emb    = (const float*)d_in[10];
  p.actor_emb   = (const float*)d_in[11];
  p.atype_emb   = (const float*)d_in[12];
  p.legal_W     = (const float*)d_in[13];
  p.legal_b     = (const float*)d_in[14];
  p.legal_g     = (const float*)d_in[15];
  p.legal_be    = (const float*)d_in[16];
  p.cls_W       = (const float*)d_in[17];
  p.cls_b       = (const float*)d_in[18];
  p.cls_g       = (const float*)d_in[19];
  p.cls_be      = (const float*)d_in[20];
  p.ctx_W       = (const float*)d_in[21];
  p.ctx_b       = (const float*)d_in[22];
  p.ctx_g       = (const float*)d_in[23];
  p.ctx_be      = (const float*)d_in[24];
  p.out         = (float*)d_out;

  const int tiles  = (BATCH * SEQ) / 16;   // 8192
  const int blocks = tiles / 8;            // 1024 (8 waves per 256-thread block)
  hipLaunchKernelGGL(poker_fused_embedding_kernel, dim3(blocks), dim3(256), 0, stream, p);
}
